// NearestNeighborEncoder_22600117911816
// MI455X (gfx1250) — matrix operand order repaired and checked by reading, not compiled, not run on hardware
//
#include <hip/hip_runtime.h>
#include <hip/hip_bf16.h>

typedef __attribute__((ext_vector_type(16))) __bf16 v16bf;
typedef __attribute__((ext_vector_type(8)))  float  v8f;
typedef unsigned short us;

#if defined(__has_builtin)
#if __has_builtin(__builtin_amdgcn_global_load_async_to_lds_b128) && \
    __has_builtin(__builtin_amdgcn_s_wait_asynccnt)
#define USE_ASYNC_LDS 1
#endif
#endif
#ifndef USE_ASYNC_LDS
#define USE_ASYNC_LDS 0
#endif

#if USE_ASYNC_LDS
typedef int v4i_t __attribute__((vector_size(16)));
typedef __attribute__((address_space(3))) v4i_t* lds_v4i_p;
#endif

// ---------------------------------------------------------------- helpers
__device__ __forceinline__ us f2bf_bits(float f) {
  union { float f; unsigned u; } cv; cv.f = f;
  unsigned r = cv.u + 0x7FFFu + ((cv.u >> 16) & 1u);
  return (us)(r >> 16);
}
__device__ __forceinline__ float bf2f(us h) {
  union { unsigned u; float f; } cv; cv.u = ((unsigned)h) << 16; return cv.f;
}

union Frag { v16bf vec; uint4 q[2]; us s[16]; };

// ---------------------------------------------------------------- layernorm (fp32 in -> bf16 out), D=768, 1 block/row
__global__ void ln_kernel(const float* __restrict__ x, const float* __restrict__ g,
                          const float* __restrict__ b, us* __restrict__ y) {
  __shared__ float red[256];
  const int tid = threadIdx.x;
  const float* xr = x + (size_t)blockIdx.x * 768;
  float vals[3]; float s = 0.f;
#pragma unroll
  for (int i = 0; i < 3; ++i) { vals[i] = xr[tid + i * 256]; s += vals[i]; }
  red[tid] = s; __syncthreads();
  for (int st = 128; st > 0; st >>= 1) { if (tid < st) red[tid] += red[tid + st]; __syncthreads(); }
  const float mu = red[0] * (1.f / 768.f);
  __syncthreads();
  float vs = 0.f;
#pragma unroll
  for (int i = 0; i < 3; ++i) { float d = vals[i] - mu; vs += d * d; }
  red[tid] = vs; __syncthreads();
  for (int st = 128; st > 0; st >>= 1) { if (tid < st) red[tid] += red[tid + st]; __syncthreads(); }
  const float inv = rsqrtf(red[0] * (1.f / 768.f) + 1e-5f);
  us* yr = y + (size_t)blockIdx.x * 768;
#pragma unroll
  for (int i = 0; i < 3; ++i) {
    int j = tid + i * 256;
    yr[j] = f2bf_bits((vals[i] - mu) * inv * g[j] + b[j]);
  }
}

// ---------------------------------------------------------------- transpose+convert: W[K,N] fp32 -> Wt[N,K] bf16
__global__ void transpose_bf16(const float* __restrict__ W, us* __restrict__ Wt, int K, int N) {
  int idx = blockIdx.x * 256 + threadIdx.x;
  if (idx >= K * N) return;
  int kk = idx / N, n = idx - kk * N;
  Wt[(size_t)n * K + kk] = f2bf_bits(W[idx]);
}

// ---------------------------------------------------------------- RoPE on q/k bf16 [8192,12,64], seq len 128
__global__ void rope_kernel(us* __restrict__ qk) {
  int gid = blockIdx.x * 256 + threadIdx.x;   // 8192*12*32 threads
  int j = gid & 31;
  int head = (gid >> 5) % 12;
  int tok = gid / (12 * 32);
  int pos = tok & 127;
  float theta = __expf(-(float)(2 * j) * (1.0f / 64.0f) * 9.210340371976184f); // 10000^(-2j/64)
  float f = (float)pos * theta;
  float c = __cosf(f), s = __sinf(f);
  size_t base = (size_t)tok * 768 + head * 64 + j;
  float x0 = bf2f(qk[base]), x1 = bf2f(qk[base + 32]);
  qk[base]      = f2bf_bits(x0 * c - x1 * s);
  qk[base + 32] = f2bf_bits(x1 * c + x0 * s);
}

// ---------------------------------------------------------------- bf16 WMMA GEMM, register+LDS blocked:
// out[M,N] = A[M,K](bf16) * Wt[N,K](bf16)^T + bias (+GELU) (+res fp32), out bf16 or fp32
// block (32,8): 8 waves. Block tile: 128(M) x 64(N). Each wave: 16 rows x 4 N-tiles,
// A-fragment reused 4x; Bt 64x32 K-slab staged in LDS (double-buffered, async fill when available).
// M % 128 == 0, N % 64 == 0, K % 32 == 0.
template<int OUT_F32, int GELU, int RES>
__global__ void gemm_kernel(const us* __restrict__ A, const us* __restrict__ Bt,
                            const float* __restrict__ bias, const float* res,
                            void* out, int M, int N, int K) {
  __shared__ us bt[2][64 * 32];                      // 2 x 4KB
  const int lane = threadIdx.x, w = threadIdx.y;
  const int hf = lane >> 4, l15 = lane & 15;
  const int tid = w * 32 + lane;
  const int mtile = blockIdx.y * 8 + w;
  const int nbase = blockIdx.x * 64;
  const us* arp = A + (size_t)(mtile * 16 + l15) * K;
  const int aoff = hf * 8;

  // cooperative Bt-slab loader: thread tid fills 8 contiguous bf16 (16B)
  const int lc = tid >> 2, lk = (tid & 3) * 8;       // col 0..63, k-chunk 0/8/16/24
  auto loadB = [&](int buf, int k0) {
    const us* src = Bt + (size_t)(nbase + lc) * K + k0 + lk;
    us* dst = &bt[buf][lc * 32 + lk];
#if USE_ASYNC_LDS
    __builtin_amdgcn_global_load_async_to_lds_b128((v4i_t*)src, (lds_v4i_p)dst, 0, 0);
#else
    *(uint4*)dst = *(const uint4*)src;
#endif
  };

  v8f acc[4] = {{}, {}, {}, {}};
  const int nk = K / 32;
  loadB(0, 0);
#if USE_ASYNC_LDS
  __builtin_amdgcn_s_wait_asynccnt(0);
#endif
  __syncthreads();
  for (int ki = 0; ki < nk; ++ki) {
    const int k0 = ki * 32;
    if (ki + 1 < nk) loadB((ki + 1) & 1, k0 + 32);   // overlap next fill with compute
    __builtin_prefetch(arp + k0 + 256, 0, 0);
    Frag fa;
    fa.q[0] = *(const uint4*)(arp + k0 + aoff);
    fa.q[1] = *(const uint4*)(arp + k0 + 16 + aoff);
    const us* bb = &bt[ki & 1][0];
#pragma unroll
    for (int t = 0; t < 4; ++t) {
      Frag fb;
      fb.q[0] = *(const uint4*)(bb + (t * 16 + l15) * 32 + hf * 8);
      fb.q[1] = *(const uint4*)(bb + (t * 16 + l15) * 32 + hf * 8 + 16);
      acc[t] = __builtin_amdgcn_wmma_f32_16x16x32_bf16(false, fa.vec, false, fb.vec,
                                                       (short)0, acc[t], false, false);
    }
#if USE_ASYNC_LDS
    __builtin_amdgcn_s_wait_asynccnt(0);
#endif
    __syncthreads();
  }

#pragma unroll
  for (int t = 0; t < 4; ++t) {
    const int col = nbase + t * 16 + l15;
    const float bv = bias[col];
#pragma unroll
    for (int vv = 0; vv < 8; ++vv) {
      int row = mtile * 16 + vv + hf * 8;
      float val = acc[t][vv] + bv;
      if (GELU) val = 0.5f * val * (1.0f + erff(val * 0.70710678118654752f));
      if (RES)  val += res[(size_t)row * N + col];
      if (OUT_F32) ((float*)out)[(size_t)row * N + col] = val;
      else         ((us*)out)[(size_t)row * N + col]    = f2bf_bits(val);
    }
  }
}

// ---------------------------------------------------------------- fused attention, one block per (seq, head)
// q [nseq*128, 768] bf16, k/v [(nseq/kv_div)*Lk, 768] bf16, o bf16. Lq=128, Lk in {64,128}.
__global__ void attn_kernel(const us* __restrict__ qp, const us* __restrict__ kp,
                            const us* __restrict__ vp, us* __restrict__ op,
                            int Lk, int kv_div, float scale) {
  __shared__ float sc[128 * 128];                       // 64 KB
  const int head = blockIdx.x % 12;
  const int seq = blockIdx.x / 12;
  const int kvseq = seq / kv_div;
  const int lane = threadIdx.x, w = threadIdx.y;
  const int hf = lane >> 4, l15 = lane & 15;
  const size_t qbase  = (size_t)(seq * 128) * 768 + head * 64;
  const size_t kvbase = (size_t)(kvseq * Lk) * 768 + head * 64;
  const int aoff = hf * 8;
  const int nct = Lk >> 4;

  // --- phase 1: scores S = scale * Q K^T, wave w owns rows [16w,16w+16)
  const us* qrow = qp + qbase + (size_t)(16 * w + l15) * 768;
  for (int ct = 0; ct < nct; ++ct) {
    v8f acc = {};
    const us* krow = kp + kvbase + (size_t)(ct * 16 + l15) * 768 + hf * 8;
#pragma unroll
    for (int k0 = 0; k0 < 64; k0 += 32) {
      Frag fa, fb;
      fa.q[0] = *(const uint4*)(qrow + k0 + aoff);
      fa.q[1] = *(const uint4*)(qrow + k0 + 16 + aoff);
      fb.q[0] = *(const uint4*)(krow + k0);
      fb.q[1] = *(const uint4*)(krow + k0 + 16);
      acc = __builtin_amdgcn_wmma_f32_16x16x32_bf16(false, fa.vec, false, fb.vec,
                                                    (short)0, acc, false, false);
    }
#pragma unroll
    for (int vv = 0; vv < 8; ++vv)
      sc[(16 * w + vv + hf * 8) * Lk + ct * 16 + l15] = acc[vv] * scale;
  }
  __syncthreads();

  // --- softmax per row (128 rows; one thread per row), bf16 result stored in-place (low 16b of slot)
  const int tid = w * 32 + lane;
  if (tid < 128) {
    float* row = sc + tid * Lk;
    float m = -1e30f;
    for (int j = 0; j < Lk; ++j) m = fmaxf(m, row[j]);
    float s = 0.f;
    for (int j = 0; j < Lk; ++j) { float e = __expf(row[j] - m); row[j] = e; s += e; }
    float inv = 1.0f / s;
    for (int j = 0; j < Lk; ++j) ((us*)(row + j))[0] = f2bf_bits(row[j] * inv);
  }
  __syncthreads();

  // --- phase 2: O = P V, wave w owns rows [16w,16w+16), cols 0..63 (4 tiles)
  const float* prow = sc + (size_t)(16 * w + l15) * Lk;
  for (int ot = 0; ot < 4; ++ot) {
    v8f acc = {};
    for (int k0 = 0; k0 < Lk; k0 += 32) {
      Frag fa, fb;
#pragma unroll
      for (int i = 0; i < 8; ++i) fa.s[i]     = *(const us*)(prow + k0 + hf * 16 + i);
#pragma unroll
      for (int i = 0; i < 8; ++i) fa.s[8 + i] = *(const us*)(prow + k0 + hf * 16 + 8 + i);
      const us* vb = vp + kvbase + (size_t)(k0 + hf * 16) * 768 + ot * 16 + l15;
#pragma unroll
      for (int i = 0; i < 16; ++i) fb.s[i] = vb[(size_t)i * 768];
      acc = __builtin_amdgcn_wmma_f32_16x16x32_bf16(false, fa.vec, false, fb.vec,
                                                    (short)0, acc, false, false);
    }
#pragma unroll
    for (int vv = 0; vv < 8; ++vv) {
      int orow = 16 * w + vv + hf * 8;
      op[qbase + (size_t)orow * 768 + ot * 16 + l15] = f2bf_bits(acc[vv]);
    }
  }
}

// ---------------------------------------------------------------- host orchestration
struct AttnP { const float *qw,*qb,*kw,*kb,*vw,*vb,*ow,*ob,*g,*b; };
struct FfnP  { const float *w1,*b1,*w2,*b2,*g,*b; };

extern "C" void kernel_launch(void* const* d_in, const int* in_sizes, int n_in,
                              void* d_out, int out_size, void* d_ws, size_t ws_size,
                              hipStream_t stream) {
  (void)n_in; (void)ws_size;
  const float* e_in = (const float*)d_in[0];
  const float* h_in = (const float*)d_in[1];
  auto F = [&](int idx) { return (const float*)d_in[idx]; };

  AttnP sa[3], ca[2]; FfnP ffn[3];
  const float *in_g, *in_b;
  if (in_sizes[2] == 589824) {
    // insertion order: sa[3]{qw,qb,kw,kb,vw,vb,ow,ob,g,b}, ca[2], ffn[3]{w1,b1,w2,b2,g,b}, in_g, in_b
    int i = 2;
    for (int l = 0; l < 3; ++l) { AttnP& p = sa[l]; p.qw=F(i++);p.qb=F(i++);p.kw=F(i++);p.kb=F(i++);p.vw=F(i++);p.vb=F(i++);p.ow=F(i++);p.ob=F(i++);p.g=F(i++);p.b=F(i++); }
    for (int l = 0; l < 2; ++l) { AttnP& p = ca[l]; p.qw=F(i++);p.qb=F(i++);p.kw=F(i++);p.kb=F(i++);p.vw=F(i++);p.vb=F(i++);p.ow=F(i++);p.ob=F(i++);p.g=F(i++);p.b=F(i++); }
    for (int l = 0; l < 3; ++l) { FfnP& p = ffn[l]; p.w1=F(i++);p.b1=F(i++);p.w2=F(i++);p.b2=F(i++);p.g=F(i++);p.b=F(i++); }
    in_g = F(i++); in_b = F(i++);
  } else {
    // jax-sorted order: ca[2]{b,g,kb,kw,ob,ow,qb,qw,vb,vw}, ffn[3]{b,b1,b2,g,w1,w2}, in_b, in_g, sa[3]
    int i = 2;
    for (int l = 0; l < 2; ++l) { AttnP& p = ca[l]; p.b=F(i++);p.g=F(i++);p.kb=F(i++);p.kw=F(i++);p.ob=F(i++);p.ow=F(i++);p.qb=F(i++);p.qw=F(i++);p.vb=F(i++);p.vw=F(i++); }
    for (int l = 0; l < 3; ++l) { FfnP& p = ffn[l]; p.b=F(i++);p.b1=F(i++);p.b2=F(i++);p.g=F(i++);p.w1=F(i++);p.w2=F(i++); }
    in_b = F(i++); in_g = F(i++);
    for (int l = 0; l < 3; ++l) { AttnP& p = sa[l]; p.b=F(i++);p.g=F(i++);p.kb=F(i++);p.kw=F(i++);p.ob=F(i++);p.ow=F(i++);p.qb=F(i++);p.qw=F(i++);p.vb=F(i++);p.vw=F(i++); }
  }

  // workspace layout
  char* wsp = (char*)d_ws;
  us* xln = (us*)wsp; wsp += (size_t)8192 * 768 * 2;
  us* wt  = (us*)wsp; wsp += (size_t)3072 * 768 * 2;
  us* qb_ = (us*)wsp; wsp += (size_t)8192 * 768 * 2;
  us* kb_ = (us*)wsp; wsp += (size_t)8192 * 768 * 2;
  us* vb_ = (us*)wsp; wsp += (size_t)8192 * 768 * 2;
  us* aob = (us*)wsp; wsp += (size_t)8192 * 768 * 2;
  us* hsb = (us*)wsp; wsp += (size_t)2048 * 768 * 2;
  us* ffh = qb_;   // FFN hidden [8192,3072] aliases q/k/v/ao region

  float* e_cur = (float*)d_out;
  (void)hipMemcpyAsync(d_out, e_in, (size_t)out_size * sizeof(float),
                       hipMemcpyDeviceToDevice, stream);

  const dim3 blk(32, 8);
  auto tr = [&](const float* W, int K, int N) {
    transpose_bf16<<<(K * N + 255) / 256, 256, 0, stream>>>(W, wt, K, N);
  };
  auto gemm_bf = [&](const us* A, const float* bias, us* out, int M, int N, int K) {
    gemm_kernel<0,0,0><<<dim3(N/64, M/128), blk, 0, stream>>>(A, wt, bias, nullptr, out, M, N, K);
  };
  auto gemm_gelu = [&](const us* A, const float* bias, us* out, int M, int N, int K) {
    gemm_kernel<0,1,0><<<dim3(N/64, M/128), blk, 0, stream>>>(A, wt, bias, nullptr, out, M, N, K);
  };
  auto gemm_res = [&](const us* A, const float* bias, float* resout, int M, int N, int K) {
    gemm_kernel<1,0,1><<<dim3(N/64, M/128), blk, 0, stream>>>(A, wt, bias, resout, resout, M, N, K);
  };

  // input layernorm of h -> hs (2048 rows)
  ln_kernel<<<2048, 256, 0, stream>>>(h_in, in_g, in_b, hsb);

  int ci = 0;
  for (int layer = 0; layer < 3; ++layer) {
    // ---- self attention
    {
      const AttnP& p = sa[layer];
      ln_kernel<<<8192, 256, 0, stream>>>(e_cur, p.g, p.b, xln);
      tr(p.qw, 768, 768); gemm_bf(xln, p.qb, qb_, 8192, 768, 768);
      tr(p.kw, 768, 768); gemm_bf(xln, p.kb, kb_, 8192, 768, 768);
      tr(p.vw, 768, 768); gemm_bf(xln, p.vb, vb_, 8192, 768, 768);
      rope_kernel<<<12288, 256, 0, stream>>>(qb_);
      rope_kernel<<<12288, 256, 0, stream>>>(kb_);
      attn_kernel<<<768, blk, 0, stream>>>(qb_, kb_, vb_, aob, 128, 1, 0.125f);
      tr(p.ow, 768, 768); gemm_res(aob, p.ob, e_cur, 8192, 768, 768);
    }
    // ---- cross attention on layers 0, 2
    if (layer == 0 || layer == 2) {
      const AttnP& p = ca[ci++];
      ln_kernel<<<8192, 256, 0, stream>>>(e_cur, p.g, p.b, xln);
      tr(p.qw, 768, 768); gemm_bf(xln, p.qb, qb_, 8192, 768, 768);
      tr(p.kw, 768, 768); gemm_bf(hsb, p.kb, kb_, 2048, 768, 768);
      tr(p.vw, 768, 768); gemm_bf(hsb, p.vb, vb_, 2048, 768, 768);
      attn_kernel<<<768, blk, 0, stream>>>(qb_, kb_, vb_, aob, 64, 2, 0.125f);
      tr(p.ow, 768, 768); gemm_res(aob, p.ob, e_cur, 8192, 768, 768);
    }
    // ---- FFN
    {
      const FfnP& f = ffn[layer];
      ln_kernel<<<8192, 256, 0, stream>>>(e_cur, f.g, f.b, xln);
      tr(f.w1, 768, 3072); gemm_gelu(xln, f.b1, ffh, 8192, 3072, 768);
      tr(f.w2, 3072, 768); gemm_res(ffh, f.b2, e_cur, 8192, 768, 3072);
    }
  }
}
